// Linear_180388626975
// MI455X (gfx1250) — matrix operand order repaired and checked by reading, not compiled, not run on hardware
//
#include <hip/hip_runtime.h>
#include <hip/hip_bf16.h>

// ---------------------------------------------------------------------------
// Decoded-weight linear layer for MI455X (gfx1250, wave32, WMMA).
//   M = 8*2048 = 16384 tokens, N = 4096 out, K = 4096 in  (549 GFLOP).
// LDS-staged, double-buffered bf16 WMMA GEMM using the CDNA5 async
// global->LDS path (ASYNCcnt) to cut L2 operand traffic 3x vs direct loads.
// x (bf16, 134MB) + w (bf16, 33.5MB) stay resident in the 192MB L2.
// ---------------------------------------------------------------------------

typedef __attribute__((ext_vector_type(16))) __bf16 v16bf;
typedef __attribute__((ext_vector_type(8)))  float  v8f;
typedef __attribute__((ext_vector_type(4)))  int    v4i;

#define OUT_F    4096
#define IN_F     4096
#define M_TOK    16384      // 8 * 2048
#define CONV_DIM 16

#define BM   128            // block tile M
#define BN   128            // block tile N
#define KS   64             // K step per stage
#define LDP  72             // padded LDS row stride (elements) -> conflict-free

union Frag {                 // one WMMA 16-bit operand: 8 VGPRs / lane
    v16bf v;
    uint4 u[2];
};

// ------------------------- async global -> LDS ------------------------------
typedef __attribute__((address_space(1))) v4i* gptr4;   // global
typedef __attribute__((address_space(3))) v4i* lptr4;   // LDS

#if __has_builtin(__builtin_amdgcn_global_load_async_to_lds_b128)
#define HAVE_ASYNC_LDS 1
__device__ __forceinline__ void async_copy16(void* l, const void* g) {
    __builtin_amdgcn_global_load_async_to_lds_b128(
        (gptr4)g, (lptr4)l, /*offset=*/0, /*cpol=*/0);
}
#else
#define HAVE_ASYNC_LDS 0
__device__ __forceinline__ void async_copy16(void* l, const void* g) {
    *(uint4*)l = *(const uint4*)g;   // synchronous fallback
}
#endif

template <int N>
__device__ __forceinline__ void wait_async() {
#if __has_builtin(__builtin_amdgcn_s_wait_asynccnt)
    __builtin_amdgcn_s_wait_asynccnt(N);
#else
    asm volatile("s_wait_asynccnt %0" ::"n"(N));
#endif
}

// ---------------------------------------------------------------------------
// Kernel 1: decode per-block latents into bf16 weight matrix w[o][i].
// ---------------------------------------------------------------------------
__global__ __launch_bounds__(256) void decode_w_kernel(
    const float* __restrict__ w_latent,   // (512*512, 16)
    const float* __restrict__ dec_w,      // (64, 16)
    const float* __restrict__ dec_b,      // (64,)
    __bf16* __restrict__ W)               // (4096, 4096) bf16 row-major
{
    int t   = blockIdx.x * 256 + threadIdx.x;
    int blk = t >> 6;
    int j   = t & 63;

    const float* lat = w_latent + (size_t)blk * CONV_DIM;
    const float* dw  = dec_w + j * CONV_DIM;
    float acc = dec_b[j];
#pragma unroll
    for (int d = 0; d < CONV_DIM; ++d) acc += lat[d] * dw[d];

    int b  = blk >> 9;
    int c  = blk & 511;
    int b1 = j >> 3;
    int c1 = j & 7;
    W[(size_t)(b * 8 + b1) * IN_F + (c * 8 + c1)] = (__bf16)acc;
}

// ---------------------------------------------------------------------------
// Kernel 2: x (f32) -> bf16, 4 elements / thread.
// ---------------------------------------------------------------------------
__global__ __launch_bounds__(256) void convert_x_kernel(
    const float4* __restrict__ X, uint2* __restrict__ Xb)
{
    int t = blockIdx.x * 256 + threadIdx.x;
    float4 f = X[t];
    union { __bf16 h[4]; uint2 u; } o;
    o.h[0] = (__bf16)f.x;
    o.h[1] = (__bf16)f.y;
    o.h[2] = (__bf16)f.z;
    o.h[3] = (__bf16)f.w;
    Xb[t] = o.u;
}

// ---------------------------------------------------------------------------
// Kernel 3: LDS-staged bf16 WMMA GEMM.
//   256 threads = 8 waves (4 M-rows x 2 N-cols), wave tile 32x64,
//   2x4 grid of 16x16 f32 accumulators, K staged 64 at a time,
//   double-buffered LDS filled with global_load_async_to_lds_b128.
//
// Fragment layouts (CDNA5 ISA 7.12.2, 16-bit operands, wave32):
//  A (16x32 MxK): lane L -> row L%16; elems 0..7 = K (L/16)*8+e,
//                 elems 8..15 = K 16+(L/16)*8+(e-8)   (2 x 16B ds loads)
//  B (32x16 KxN): lane L -> col L%16; same K mapping as A (2 x 16B)
// ---------------------------------------------------------------------------
__global__ __launch_bounds__(256) void gemm_bf16_kernel(
    const __bf16* __restrict__ Xb,    // M x K row-major
    const __bf16* __restrict__ Wb,    // N x K row-major
    const float*  __restrict__ bias,  // N
    float* __restrict__ Out)          // M x N
{
    const int K = IN_F;
    const int N = OUT_F;

    __shared__ __bf16 sA[2][BM * LDP];   // 2 x 18 KB
    __shared__ __bf16 sB[2][BN * LDP];   // 2 x 18 KB

    const int tid  = threadIdx.x;
    const int lane = tid & 31;
    const int wave = tid >> 5;           // 0..7
    const int wr   = wave & 3;           // wave M row (0..3) -> 32 rows each
    const int wc   = wave >> 2;          // wave N col (0..1) -> 64 cols each

    const int rowA0 = blockIdx.y * BM;   // block M origin
    const int colB0 = blockIdx.x * BN;   // block N origin

    const int lr = lane & 15;            // row/col within 16x16 tile
    const int lh = lane >> 4;            // half-select

    v8f acc[2][4] = {};

    // ---- cooperative stage: 1024 16B chunks per matrix, 4 per thread ------
    auto stage = [&](int buf, int k0) {
#pragma unroll
        for (int i = 0; i < 4; ++i) {
            const int c   = tid + i * 256;     // 0..1023
            const int row = c >> 3;            // 128 rows
            const int ce  = (c & 7) * 8;       // element offset in K-slab
            async_copy16(&sA[buf][row * LDP + ce],
                         Xb + (size_t)(rowA0 + row) * K + k0 + ce);
            async_copy16(&sB[buf][row * LDP + ce],
                         Wb + (size_t)(colB0 + row) * K + k0 + ce);
        }
    };

    // per-lane LDS fragment base offsets (elements)
    const int aoff = (wr * 32 + lr) * LDP + lh * 8;    // + mi*16*LDP + kk
    const int boff = (wc * 64 + lr) * LDP + lh * 8;    // + ni*16*LDP + kk

    const int NKB = K / KS;              // 64 stages
    stage(0, 0);

    for (int kb = 0; kb < NKB; ++kb) {
        const int buf = kb & 1;
        if (kb + 1 < NKB) {
            stage(buf ^ 1, (kb + 1) * KS);
            wait_async<8>();             // current buf's 8 copies done
        } else {
            wait_async<0>();
        }
        __syncthreads();                 // all waves' copies visible

#pragma unroll
        for (int ksub = 0; ksub < 2; ++ksub) {
            const int kk = ksub * 32;
            Frag a[2], b[4];
#pragma unroll
            for (int mi = 0; mi < 2; ++mi) {
                const __bf16* p = &sA[buf][aoff + mi * 16 * LDP + kk];
                a[mi].u[0] = *(const uint4*)(p);
                a[mi].u[1] = *(const uint4*)(p + 16);
            }
#pragma unroll
            for (int ni = 0; ni < 4; ++ni) {
                const __bf16* p = &sB[buf][boff + ni * 16 * LDP + kk];
                b[ni].u[0] = *(const uint4*)(p);
                b[ni].u[1] = *(const uint4*)(p + 16);
            }
#pragma unroll
            for (int mi = 0; mi < 2; ++mi)
#pragma unroll
                for (int ni = 0; ni < 4; ++ni)
                    acc[mi][ni] = __builtin_amdgcn_wmma_f32_16x16x32_bf16(
                        false, a[mi].v, false, b[ni].v,
                        (short)0, acc[mi][ni], false, false);
        }
        __syncthreads();                 // protect buf from next-next stage
    }

    // C/D layout: lane L, VGPR j -> n = L%16, m = (L/16)*8 + j
#pragma unroll
    for (int ni = 0; ni < 4; ++ni) {
        const int n  = colB0 + wc * 64 + ni * 16 + lr;
        const float bv = bias[n];
#pragma unroll
        for (int mi = 0; mi < 2; ++mi) {
            const int mbase = rowA0 + wr * 32 + mi * 16 + lh * 8;
#pragma unroll
            for (int j = 0; j < 8; ++j)
                Out[(size_t)(mbase + j) * N + n] = acc[mi][ni][j] + bv;
        }
    }
}

// ---------------------------------------------------------------------------
extern "C" void kernel_launch(void* const* d_in, const int* in_sizes, int n_in,
                              void* d_out, int out_size, void* d_ws, size_t ws_size,
                              hipStream_t stream) {
    const float* x        = (const float*)d_in[0];   // (8,2048,4096)
    const float* w_latent = (const float*)d_in[1];   // (262144,16)
    const float* dec_w    = (const float*)d_in[2];   // (64,16)
    const float* dec_b    = (const float*)d_in[3];   // (64,)
    const float* bias     = (const float*)d_in[4];   // (4096,)
    float* out = (float*)d_out;                      // (8,2048,4096)

    __bf16* Xbuf = (__bf16*)d_ws;                                        // 134.2 MB
    __bf16* Wbuf = (__bf16*)((char*)d_ws + (size_t)M_TOK * IN_F * 2);    //  33.6 MB

    decode_w_kernel<<<(512 * 512 * 64) / 256, 256, 0, stream>>>(
        w_latent, dec_w, dec_b, Wbuf);

    convert_x_kernel<<<((size_t)M_TOK * IN_F / 4) / 256, 256, 0, stream>>>(
        (const float4*)x, (uint2*)Xbuf);

    dim3 grid(OUT_F / BN, M_TOK / BM);   // (32, 128)
    gemm_bf16_kernel<<<grid, 256, 0, stream>>>(Xbuf, Wbuf, bias, out);
}
